// RelPartialLearnableMultiHeadAttn_1975684956269
// MI455X (gfx1250) — matrix operand order repaired and checked by reading, not compiled, not run on hardware
//
#include <hip/hip_runtime.h>
#include <hip/hip_bf16.h>
#include <stdint.h>

#define NH    16
#define DH    64
#define DM    1024
#define QL    1024
#define MEML  1024
#define KLN   2048
#define BS    2
#define RKROWS (KLN + 96)   // pad so rel-shift band reads never go OOB

typedef __attribute__((ext_vector_type(16))) __bf16 bf16x16;
typedef __attribute__((ext_vector_type(8)))  float  f32x8;
typedef unsigned int u32x4 __attribute__((ext_vector_type(4)));
typedef int          i32x8 __attribute__((ext_vector_type(8)));
typedef int          i32x4 __attribute__((ext_vector_type(4)));

union Frag {
  bf16x16 v;
  uint4   u[2];
};

__device__ __forceinline__ unsigned short f2bf(float f) {
  unsigned int u = __float_as_uint(f);
  u += 0x7FFFu + ((u >> 16) & 1u);
  return (unsigned short)(u >> 16);
}

__device__ __forceinline__ f32x8 wmma_bf16(Frag a, Frag b, f32x8 c) {
  return __builtin_amdgcn_wmma_f32_16x16x32_bf16(false, a.v, false, b.v,
                                                 (short)0, c, false, false);
}

__device__ __forceinline__ uint32_t lds_off(const void* p) {
  return (uint32_t)(uintptr_t)p;   // flat addr low 32 bits == LDS offset
}

// TDM 2-D tile load: tile_h rows x tile_w bf16 elements, row stride in elements.
// D# packing per CDNA5 ISA (groups 0/1; 2/3 zero for 2D).
// amdgpu-toolchain (clang-23) builtin form: 6 args (g0, g1, g2, g3, g4, cpol).
__device__ __forceinline__ void tdm_load_2d(uint32_t lds, const void* gsrc,
                                            uint32_t tile_w, uint32_t tile_h,
                                            uint32_t stride_elems) {
  uint64_t ga = (uint64_t)(uintptr_t)gsrc;
  u32x4 g0;
  g0[0] = 1u;                                   // count=1 (valid user descriptor)
  g0[1] = lds;                                  // lds_addr (bytes)
  g0[2] = (uint32_t)ga;                         // global_addr[31:0]
  g0[3] = (uint32_t)(ga >> 32) | (2u << 30);    // global_addr[56:32] | type=2
  i32x8 g1;
  g1[0] = (int)(1u << 16);                                      // data_size=1 (2B)
  g1[1] = (int)((tile_w & 0xFFFFu) << 16);                      // tensor_dim0 lo
  g1[2] = (int)(((tile_w >> 16) & 0xFFFFu) |
                ((tile_h & 0xFFFFu) << 16));                    // td0 hi | tensor_dim1 lo
  g1[3] = (int)(((tile_h >> 16) & 0xFFFFu) | (tile_w << 16));   // td1 hi | tile_dim0
  g1[4] = (int)(tile_h & 0xFFFFu);                              // tile_dim1 | tile_dim2=0
  g1[5] = (int)stride_elems;                                    // tensor_dim0_stride lo
  g1[6] = 0;
  g1[7] = 0;
  i32x4 gz4 = {0, 0, 0, 0};
  i32x8 gz8 = {0, 0, 0, 0, 0, 0, 0, 0};
  __builtin_amdgcn_tensor_load_to_lds(g0, g1, gz4, gz4, gz8, 0);
}

// ---------------------------------------------------------------- converts
__global__ __launch_bounds__(256) void f2bf_kernel(const float* __restrict__ s,
                                                   unsigned short* __restrict__ d, int n) {
  int i = blockIdx.x * 256 + threadIdx.x;
  if (i < n) d[i] = f2bf(s[i]);
}

__global__ __launch_bounds__(256) void build_cat_kernel(const float* __restrict__ mems,
                                                        const float* __restrict__ w,
                                                        unsigned short* __restrict__ cat) {
  int i = blockIdx.x * 256 + threadIdx.x;          // over 4096*1024
  const int half = KLN * DM;                       // mems part: rows t<1024 (row=t*2+b)
  if (i < half) cat[i] = f2bf(mems[i]);
  else          cat[i] = f2bf(w[i - half]);
}

__global__ __launch_bounds__(256) void zero_us_kernel(unsigned short* __restrict__ d, int n) {
  int i = blockIdx.x * 256 + threadIdx.x;
  if (i < n) d[i] = 0;
}

// ---------------------------------------------------------------- GEMM
// C[m,n] = sum_k A[m,k] * B[n,k], A/B bf16 row-major (ld = K), fp32 accum.
// Tiles staged via TDM (double-buffered). MODE 0: QKV split; 1: bf16; 2: f32.
template <int MODE>
__global__ __launch_bounds__(256) void gemm_bf16_kernel(
    const unsigned short* __restrict__ A, const unsigned short* __restrict__ Bm,
    int M, int N, int K,
    float* __restrict__ outF, unsigned short* __restrict__ outB,
    unsigned short* __restrict__ qw, unsigned short* __restrict__ qr,
    unsigned short* __restrict__ kb, unsigned short* __restrict__ vt,
    const float* __restrict__ rwb, const float* __restrict__ rrb) {
  const int BM = 128, BN = 128, BK = 32;
  __shared__ unsigned short As[2][BM * BK];
  __shared__ unsigned short Bs[2][BN * BK];

  const int tid  = threadIdx.x;
  const int lane = tid & 31;
  const int wave = tid >> 5;           // 0..7
  const int wm   = wave & 3;           // m offset 32*wm
  const int wn   = wave >> 2;          // n offset 64*wn
  const int n15  = lane & 15;
  const int kh   = lane >> 4;
  const int bm0  = blockIdx.x * BM;
  const int bn0  = blockIdx.y * BN;

  f32x8 acc[2][4];
  f32x8 zero = {0.f, 0.f, 0.f, 0.f, 0.f, 0.f, 0.f, 0.f};
#pragma unroll
  for (int mi = 0; mi < 2; ++mi)
#pragma unroll
    for (int ni = 0; ni < 4; ++ni) acc[mi][ni] = zero;

  if (wave == 0) {  // prologue: stage k0=0 tiles into buffer 0
    tdm_load_2d(lds_off(&As[0][0]), A + (size_t)bm0 * K, BK, BM, K);
    tdm_load_2d(lds_off(&Bs[0][0]), Bm + (size_t)bn0 * K, BK, BN, K);
  }

  int buf = 0;
  for (int k0 = 0; k0 < K; k0 += BK) {
    if (wave == 0) {
      if (k0 + BK < K) {  // prefetch next tiles into other buffer
        tdm_load_2d(lds_off(&As[buf ^ 1][0]), A + (size_t)bm0 * K + k0 + BK, BK, BM, K);
        tdm_load_2d(lds_off(&Bs[buf ^ 1][0]), Bm + (size_t)bn0 * K + k0 + BK, BK, BN, K);
        __builtin_amdgcn_s_wait_tensorcnt(2);  // current buffer's 2 DMAs retired
      } else {
        __builtin_amdgcn_s_wait_tensorcnt(0);
      }
    }
    __syncthreads();  // LDS tiles ready for all waves

    Frag af[2], bf[4];
#pragma unroll
    for (int mi = 0; mi < 2; ++mi) {
      int rm = wm * 32 + mi * 16 + n15;
      af[mi].u[0] = *(const uint4*)&As[buf][rm * BK + kh * 16];
      af[mi].u[1] = *(const uint4*)&As[buf][rm * BK + kh * 16 + 8];
    }
#pragma unroll
    for (int ni = 0; ni < 4; ++ni) {
      int rn = wn * 64 + ni * 16 + n15;
      bf[ni].u[0] = *(const uint4*)&Bs[buf][rn * BK + kh * 16];
      bf[ni].u[1] = *(const uint4*)&Bs[buf][rn * BK + kh * 16 + 8];
    }
#pragma unroll
    for (int mi = 0; mi < 2; ++mi)
#pragma unroll
      for (int ni = 0; ni < 4; ++ni)
        acc[mi][ni] = wmma_bf16(af[mi], bf[ni], acc[mi][ni]);

    __syncthreads();  // all waves done with buf before it is re-filled
    buf ^= 1;
  }

#pragma unroll
  for (int mi = 0; mi < 2; ++mi) {
#pragma unroll
    for (int ni = 0; ni < 4; ++ni) {
      int mg0 = bm0 + wm * 32 + mi * 16 + kh * 8;
      int ng  = bn0 + wn * 64 + ni * 16 + n15;
#pragma unroll
      for (int v = 0; v < 8; ++v) {
        int mg = mg0 + v;
        float val = acc[mi][ni][v];
        if (MODE == 2) {
          outF[(size_t)mg * N + ng] = val;
        } else if (MODE == 1) {
          outB[(size_t)mg * N + ng] = f2bf(val);
        } else {  // QKV: row = t*2+b over cat, col = qkv
          int t = mg >> 1, bb = mg & 1;
          if (ng < DM) {
            if (t >= MEML) {
              int row = (t - MEML) * BS + bb;
              qw[(size_t)row * DM + ng] = f2bf(val + rwb[ng]);
              qr[(size_t)row * DM + ng] = f2bf(val + rrb[ng]);
            }
          } else if (ng < 2 * DM) {
            kb[(size_t)mg * DM + (ng - DM)] = f2bf(val);
          } else {
            int c = ng - 2 * DM;
            int h = c >> 6, d = c & 63;
            vt[(((size_t)bb * NH + h) * DH + d) * KLN + t] = f2bf(val);
          }
        }
      }
    }
  }
}

// ---------------------------------------------------------------- attention
// grid: (QL/64, BS*NH); block: 128 threads = 4 waves, 16 q-rows each.
// K/V chunks staged by wave-0 TDM (block-shared); rel-pos band staged by a
// per-wave TDM into a private LDS region (no barrier needed). All double-
// buffered so every DMA overlaps compute; all WMMA B operands come from LDS.
__global__ __launch_bounds__(128) void attn_kernel(
    const unsigned short* __restrict__ qw, const unsigned short* __restrict__ qr,
    const unsigned short* __restrict__ kb, const unsigned short* __restrict__ vt,
    const unsigned short* __restrict__ rk, unsigned short* __restrict__ attnV) {
  const float scale = 0.125f;  // 1/sqrt(64)
  const int b = blockIdx.y & 1;
  const int h = blockIdx.y >> 1;
  const int qi0 = blockIdx.x * 64;
  const int wave = threadIdx.x >> 5;
  const int lane = threadIdx.x & 31;
  const int i0 = qi0 + wave * 16;
  const int n15 = lane & 15, kh = lane >> 4;

  __shared__ unsigned short Ks[2][64 * 64];      // [j_local][d]
  __shared__ unsigned short Vs[2][64 * 64];      // [d_local][j]
  __shared__ unsigned short Rb[4][2][96 * 64];   // per-wave rel-pos band [jr_local][d]
  __shared__ float bd[4][16][96];
  __shared__ unsigned short ps[4][16][64];

  const unsigned short* kbase = kb + ((size_t)b) * DM + (size_t)h * DH;   // + j*BS*DM
  const unsigned short* vbase = vt + (((size_t)b * NH + h) * DH) * KLN;   // + d*KLN + j
  const unsigned short* rbase = rk + (size_t)h * DH;                      // + jr*DM

  // Q fragments (A layout, 2 k-steps over head dim)
  Frag aqw[2], aqr[2];
  const unsigned short* qwp = qw + ((size_t)(i0 + n15) * BS + b) * DM + h * DH;
  const unsigned short* qrp = qr + ((size_t)(i0 + n15) * BS + b) * DM + h * DH;
#pragma unroll
  for (int s = 0; s < 2; ++s) {
    aqw[s].u[0] = *(const uint4*)(qwp + s * 32 + kh * 16);
    aqw[s].u[1] = *(const uint4*)(qwp + s * 32 + kh * 16 + 8);
    aqr[s].u[0] = *(const uint4*)(qrp + s * 32 + kh * 16);
    aqr[s].u[1] = *(const uint4*)(qrp + s * 32 + kh * 16 + 8);
  }

  f32x8 zero = {0.f, 0.f, 0.f, 0.f, 0.f, 0.f, 0.f, 0.f};
  f32x8 o[4];
#pragma unroll
  for (int dt = 0; dt < 4; ++dt) o[dt] = zero;
  float mrun[8], lrun[8];
#pragma unroll
  for (int v = 0; v < 8; ++v) { mrun[v] = -1e30f; lrun[v] = 0.f; }

  const int nchunk = (qi0 >> 6) + 17;  // covers j <= qi0+63+1024 (same for all waves)

  // prologue: stage chunk 0 (wave0: K/V; every wave: its own rel-pos band)
  if (wave == 0) {
    tdm_load_2d(lds_off(&Ks[0][0]), kbase, 64, 64, BS * DM);
    tdm_load_2d(lds_off(&Vs[0][0]), vbase, 64, 64, KLN);
  }
  {
    const int jrt00 = (0 - i0 + 1008) & ~15;
    tdm_load_2d(lds_off(&Rb[wave][0][0]), rbase + (size_t)jrt00 * DM, 64, 96, DM);
  }

  int buf = 0;
  for (int c = 0; c < nchunk; ++c) {
    const int j0 = c * 64;
    const int jrt0 = (j0 - i0 + 1008) & ~15;   // >= 0; matches staged band
    if (c + 1 < nchunk) {
      const int jn = j0 + 64;
      if (wave == 0) {
        tdm_load_2d(lds_off(&Ks[buf ^ 1][0]), kbase + (size_t)jn * BS * DM, 64, 64, BS * DM);
        tdm_load_2d(lds_off(&Vs[buf ^ 1][0]), vbase + jn, 64, 64, KLN);
      }
      const int jrtn = (jn - i0 + 1008) & ~15;
      tdm_load_2d(lds_off(&Rb[wave][buf ^ 1][0]), rbase + (size_t)jrtn * DM, 64, 96, DM);
      if (wave == 0) __builtin_amdgcn_s_wait_tensorcnt(3);  // current K/V/R retired
      else           __builtin_amdgcn_s_wait_tensorcnt(1);  // current R retired
    } else {
      __builtin_amdgcn_s_wait_tensorcnt(0);
    }
    __syncthreads();  // K/V chunk ready block-wide

    // --- AC = (q+rw_bias) . K^T  (B-frags from LDS, batched per step) ----
    f32x8 sacc[4];
#pragma unroll
    for (int nt = 0; nt < 4; ++nt) sacc[nt] = zero;
#pragma unroll
    for (int s = 0; s < 2; ++s) {
      Frag bk[4];
#pragma unroll
      for (int nt = 0; nt < 4; ++nt) {
        const unsigned short* kp = &Ks[buf][(nt * 16 + n15) * 64 + s * 32 + kh * 16];
        bk[nt].u[0] = ((const uint4*)kp)[0];
        bk[nt].u[1] = *(const uint4*)(kp + 8);
      }
#pragma unroll
      for (int nt = 0; nt < 4; ++nt) sacc[nt] = wmma_bf16(aqw[s], bk[nt], sacc[nt]);
    }

    // --- BD band tiles (B-frags from per-wave LDS band) -> bd LDS ---------
#pragma unroll
    for (int tt = 0; tt < 6; ++tt) {
      Frag br[2];
#pragma unroll
      for (int s = 0; s < 2; ++s) {
        const unsigned short* rp = &Rb[wave][buf][(tt * 16 + n15) * 64 + s * 32 + kh * 16];
        br[s].u[0] = ((const uint4*)rp)[0];
        br[s].u[1] = *(const uint4*)(rp + 8);
      }
      f32x8 bacc = zero;
#pragma unroll
      for (int s = 0; s < 2; ++s) bacc = wmma_bf16(aqr[s], br[s], bacc);
#pragma unroll
      for (int v = 0; v < 8; ++v)
        bd[wave][kh * 8 + v][tt * 16 + n15] = bacc[v];
    }

    // --- merge, mask, online softmax -------------------------------------
#pragma unroll
    for (int v = 0; v < 8; ++v) {
      const int m = kh * 8 + v;
      const int irow = i0 + m;
      float sv[4];
      float mx = -1e30f;
#pragma unroll
      for (int nt = 0; nt < 4; ++nt) {
        int jc = nt * 16 + n15;
        int j = j0 + jc;
        int jroff = j - irow + 1023 - jrt0;          // in [0,96)
        float sc = (sacc[nt][v] + bd[wave][m][jroff]) * scale;
        if (j > irow + MEML) sc = -1e30f;            // attn mask
        sv[nt] = sc;
        mx = fmaxf(mx, sc);
      }
#pragma unroll
      for (int d = 1; d < 16; d <<= 1) mx = fmaxf(mx, __shfl_xor(mx, d, 32));
      float mnew = fmaxf(mrun[v], mx);
      float alpha = __expf(mrun[v] - mnew);
      float rowsum = 0.f;
#pragma unroll
      for (int nt = 0; nt < 4; ++nt) {
        float p = __expf(sv[nt] - mnew);
        rowsum += p;
        ps[wave][m][nt * 16 + n15] = f2bf(p);
      }
#pragma unroll
      for (int d = 1; d < 16; d <<= 1) rowsum += __shfl_xor(rowsum, d, 32);
      lrun[v] = lrun[v] * alpha + rowsum;
      mrun[v] = mnew;
#pragma unroll
      for (int dt = 0; dt < 4; ++dt) o[dt][v] = o[dt][v] * alpha;
    }

    // --- PV (B-frags from LDS V, batched per step) ------------------------
#pragma unroll
    for (int s = 0; s < 2; ++s) {
      Frag ap;
      const unsigned short* pp = &ps[wave][n15][0];
      ap.u[0] = *(const uint4*)(pp + s * 32 + kh * 16);
      ap.u[1] = *(const uint4*)(pp + s * 32 + kh * 16 + 8);
      Frag bv[4];
#pragma unroll
      for (int dt = 0; dt < 4; ++dt) {
        const unsigned short* vp = &Vs[buf][(dt * 16 + n15) * 64 + s * 32 + kh * 16];
        bv[dt].u[0] = ((const uint4*)vp)[0];
        bv[dt].u[1] = *(const uint4*)(vp + 8);
      }
#pragma unroll
      for (int dt = 0; dt < 4; ++dt) o[dt] = wmma_bf16(ap, bv[dt], o[dt]);
    }

    __syncthreads();  // all waves done with buf before it is re-filled
    buf ^= 1;
  }

  // --- finalize ----------------------------------------------------------
#pragma unroll
  for (int v = 0; v < 8; ++v) {
    const int m = kh * 8 + v;
    const int irow = i0 + m;
    float inv = 1.0f / lrun[v];
#pragma unroll
    for (int dt = 0; dt < 4; ++dt)
      attnV[((size_t)irow * BS + b) * DM + h * DH + dt * 16 + n15] = f2bf(o[dt][v] * inv);
  }
}

// ---------------------------------------------------------------- layernorm
__global__ __launch_bounds__(256) void ln_kernel(const float* __restrict__ w,
                                                 const float* __restrict__ ao,
                                                 const float* __restrict__ g,
                                                 const float* __restrict__ bb,
                                                 float* __restrict__ out) {
  const int row = blockIdx.x;
  const int tid = threadIdx.x;
  __shared__ float rs[8], rs2[8];
  const float* wr = w + (size_t)row * DM;
  const float* ar = ao + (size_t)row * DM;
  float x[4], s = 0.f, s2 = 0.f;
#pragma unroll
  for (int e = 0; e < 4; ++e) {
    int d = tid + e * 256;
    float t = wr[d] + ar[d];
    x[e] = t; s += t; s2 += t * t;
  }
#pragma unroll
  for (int m = 1; m < 32; m <<= 1) { s += __shfl_xor(s, m, 32); s2 += __shfl_xor(s2, m, 32); }
  int wv = tid >> 5, ln = tid & 31;
  if (ln == 0) { rs[wv] = s; rs2[wv] = s2; }
  __syncthreads();
  s = 0.f; s2 = 0.f;
#pragma unroll
  for (int i = 0; i < 8; ++i) { s += rs[i]; s2 += rs2[i]; }
  float mu = s * (1.0f / DM);
  float var = s2 * (1.0f / DM) - mu * mu;
  float rstd = rsqrtf(var + 1e-5f);
#pragma unroll
  for (int e = 0; e < 4; ++e) {
    int d = tid + e * 256;
    out[(size_t)row * DM + d] = (x[e] - mu) * rstd * g[d] + bb[d];
  }
}

// ---------------------------------------------------------------- launch
extern "C" void kernel_launch(void* const* d_in, const int* in_sizes, int n_in,
                              void* d_out, int out_size, void* d_ws, size_t ws_size,
                              hipStream_t stream) {
  (void)in_sizes; (void)n_in; (void)out_size; (void)ws_size;
  const float* w    = (const float*)d_in[0];
  const float* r    = (const float*)d_in[1];
  const float* rwb  = (const float*)d_in[2];
  const float* rrb  = (const float*)d_in[3];
  const float* mems = (const float*)d_in[4];
  /* d_in[5] attn_mask: computed analytically on device */
  const float* qkvW = (const float*)d_in[6];
  const float* rW   = (const float*)d_in[7];
  const float* oW   = (const float*)d_in[8];
  const float* lng  = (const float*)d_in[9];
  const float* lnb  = (const float*)d_in[10];
  float* out = (float*)d_out;

  char* p = (char*)d_ws;
  auto alloc = [&](size_t bytes) -> char* {
    char* q = p;
    p += (bytes + 255) & ~(size_t)255;
    return q;
  };
  unsigned short* catB  = (unsigned short*)alloc((size_t)KLN * BS * DM * 2);
  unsigned short* wqkvB = (unsigned short*)alloc((size_t)3 * DM * DM * 2);
  unsigned short* rB    = (unsigned short*)alloc((size_t)KLN * DM * 2);
  unsigned short* rWB   = (unsigned short*)alloc((size_t)DM * DM * 2);
  unsigned short* oWB   = (unsigned short*)alloc((size_t)DM * DM * 2);
  unsigned short* qwB   = (unsigned short*)alloc((size_t)QL * BS * DM * 2);
  unsigned short* qrB   = (unsigned short*)alloc((size_t)QL * BS * DM * 2);
  unsigned short* kB    = (unsigned short*)alloc((size_t)KLN * BS * DM * 2);
  unsigned short* vT    = (unsigned short*)alloc((size_t)BS * NH * DH * KLN * 2);
  unsigned short* rkB   = (unsigned short*)alloc((size_t)RKROWS * DM * 2);
  unsigned short* attnV = (unsigned short*)alloc((size_t)QL * BS * DM * 2);
  float*          aOut  = (float*)alloc((size_t)QL * BS * DM * 4);

  // converts
  f2bf_kernel<<<(3 * DM * DM + 255) / 256, 256, 0, stream>>>(qkvW, wqkvB, 3 * DM * DM);
  f2bf_kernel<<<(DM * DM + 255) / 256, 256, 0, stream>>>(rW, rWB, DM * DM);
  f2bf_kernel<<<(DM * DM + 255) / 256, 256, 0, stream>>>(oW, oWB, DM * DM);
  f2bf_kernel<<<(KLN * DM + 255) / 256, 256, 0, stream>>>(r, rB, KLN * DM);
  build_cat_kernel<<<(KLN * BS * DM) / 256, 256, 0, stream>>>(mems, w, catB);
  zero_us_kernel<<<(96 * DM) / 256, 256, 0, stream>>>(rkB + (size_t)KLN * DM, 96 * DM);

  // QKV projection: M=4096, N=3072, K=1024 (split epilogue)
  dim3 g1(KLN * BS / 128, 3 * DM / 128);
  gemm_bf16_kernel<0><<<g1, 256, 0, stream>>>(catB, wqkvB, KLN * BS, 3 * DM, DM,
                                              nullptr, nullptr, qwB, qrB, kB, vT, rwb, rrb);
  // r_head_k: M=2048, N=1024, K=1024
  dim3 g2(KLN / 128, DM / 128);
  gemm_bf16_kernel<1><<<g2, 256, 0, stream>>>(rB, rWB, KLN, DM, DM,
                                              nullptr, rkB, nullptr, nullptr, nullptr,
                                              nullptr, nullptr, nullptr);
  // attention
  dim3 ga(QL / 64, BS * NH);
  attn_kernel<<<ga, 128, 0, stream>>>(qwB, qrB, kB, vT, rkB, attnV);
  // output projection: M=2048, N=1024, K=1024
  dim3 g3(QL * BS / 128, DM / 128);
  gemm_bf16_kernel<2><<<g3, 256, 0, stream>>>(attnV, oWB, QL * BS, DM, DM,
                                              aOut, nullptr, nullptr, nullptr, nullptr,
                                              nullptr, nullptr, nullptr);
  // residual + layernorm
  ln_kernel<<<QL * BS, 256, 0, stream>>>(w, aOut, lng, lnb, out);
}
